// MPNEncoder_19327352832402
// MI455X (gfx1250) — matrix operand order repaired and checked by reading, not compiled, not run on hardware
//
#include <hip/hip_runtime.h>

// ---------------------------------------------------------------------------
// MPN GRU encoder for gfx1250 (MI455X): bf16 WMMA (16x16x32) with f32 accum.
// Gathers routed through the CDNA5 async global->LDS engine (ASYNCcnt).
// Working set (~163MB bf16) fits the 192MB L2; gathers are L2-resident.
// ---------------------------------------------------------------------------

typedef __bf16 bf16_t;
typedef __attribute__((ext_vector_type(16))) __bf16 v16bf;
typedef __attribute__((ext_vector_type(8)))  __bf16 v8bf;
typedef __attribute__((ext_vector_type(8)))  float  v8f;

#define N_NODES 50000
#define N_MESS  150000
#define MAX_NB  6
#define FD_NODE 128
#define FD_MESS 160
#define HDIM    128
#define DEPTH   5

// ---- async global->LDS copy (16B per lane), with safe fallback -------------
#if defined(__has_builtin)
#  if __has_builtin(__builtin_amdgcn_global_load_async_to_lds_b128)
#    define ASYNC_LDS 1
#  endif
#endif
#ifndef ASYNC_LDS
#  define ASYNC_LDS 0
#endif

#if ASYNC_LDS
typedef int v4i_ __attribute__((__vector_size__(16)));
typedef __attribute__((address_space(1))) v4i_* g_v4i_p;   // global (AS1)
typedef __attribute__((address_space(3))) v4i_* l_v4i_p;   // LDS (AS3)
#endif

__device__ __forceinline__ void cp16_g2lds(const void* g, void* l) {
#if ASYNC_LDS
    __builtin_amdgcn_global_load_async_to_lds_b128(
        (g_v4i_p)(uintptr_t)g,
        (l_v4i_p)(unsigned)(uintptr_t)l,
        0, 0);
#else
    *(v8bf*)l = *(const v8bf*)g;
#endif
}

__device__ __forceinline__ void async_fence() {
#if ASYNC_LDS
#  if defined(__has_builtin) && __has_builtin(__builtin_amdgcn_s_wait_asynccnt)
    __builtin_amdgcn_s_wait_asynccnt(0);
#  else
    asm volatile("s_wait_asynccnt 0x0" ::: "memory");
#  endif
#endif
}

__device__ __forceinline__ float sigmoid_f(float x) {
    return 1.0f / (1.0f + __expf(-x));
}

// A fragment (16x32 bf16, M x K): lane l holds row (l&15); K elements
// [16*kh, 16*kh+16) contiguous where kh = l>>4 (matches B fragment).
__device__ __forceinline__ v16bf load_a_frag(const bf16_t* row_kb, int kh) {
    v8bf lo = *(const v8bf*)(row_kb + 16 * kh);
    v8bf hi = *(const v8bf*)(row_kb + 16 * kh + 8);
    v16bf r;
#pragma unroll
    for (int i = 0; i < 8; ++i) { r[i] = lo[i]; r[8 + i] = hi[i]; }
    return r;
}

// B fragment (32x16 bf16, K x N) from transposed weights Wt[N][K]:
// lane l holds col (l&15), K range [16*(l>>4), +16) contiguous.
__device__ __forceinline__ v16bf load_b_frag(const bf16_t* Wt, int K,
                                             int col, int kb, int kh) {
    return *(const v16bf*)(Wt + (size_t)col * K + kb + 16 * kh);
}

__device__ __forceinline__ v8f wmma_bf16(v16bf a, v16bf b, v8f c) {
    return __builtin_amdgcn_wmma_f32_16x16x32_bf16(
        false, a, false, b, (short)0, c, false, false);
}

// --------------------------- prep kernels ----------------------------------

// W [K][128] f32 -> Wt [128][K] bf16 (transpose so B-frag loads are contiguous)
__global__ void transpose_w_kernel(const float* __restrict__ W,
                                   bf16_t* __restrict__ Wt, int K) {
    int idx = blockIdx.x * blockDim.x + threadIdx.x;
    if (idx >= K * 128) return;
    int k = idx >> 7, n = idx & 127;
    Wt[(size_t)n * K + k] = (bf16_t)W[(size_t)k * 128 + n];
}

__global__ void f32_to_bf16_kernel(const float* __restrict__ x,
                                   bf16_t* __restrict__ y, int n) {
    int idx = blockIdx.x * blockDim.x + threadIdx.x;
    if (idx < n) y[idx] = (bf16_t)x[idx];
}

__global__ void zero_bf16_kernel(bf16_t* __restrict__ y, int n) {
    int idx = blockIdx.x * blockDim.x + threadIdx.x;
    if (idx < n) y[idx] = (bf16_t)0.0f;
}

// r1 = fmess @ W_r  -> bf16 [M,128]. Block = 8 waves, 16-row tile, wave owns
// a 16-col slice. K = 160 -> 5 WMMA steps.
__global__ __launch_bounds__(256)
void r1_kernel(const bf16_t* __restrict__ fmess_bf,
               const bf16_t* __restrict__ Wrt,
               bf16_t* __restrict__ r1_bf) {
    const int tid  = threadIdx.x;
    const int wave = tid >> 5, lane = tid & 31;
    const int m0   = blockIdx.x * 16;
    const int col  = wave * 16 + (lane & 15);
    const int rl   = lane & 15, kh = lane >> 4;

    v8f acc = {0.f, 0.f, 0.f, 0.f, 0.f, 0.f, 0.f, 0.f};
#pragma unroll
    for (int kb = 0; kb < FD_MESS; kb += 32) {
        v16bf a = load_a_frag(fmess_bf + (size_t)(m0 + rl) * FD_MESS + kb, kh);
        v16bf b = load_b_frag(Wrt, FD_MESS, col, kb, kh);
        acc = wmma_bf16(a, b, acc);
    }
#pragma unroll
    for (int i = 0; i < 8; ++i)
        r1_bf[(size_t)(m0 + i + 8 * kh) * HDIM + col] = (bf16_t)acc[i];
}

// --------------------------- fused GRU step --------------------------------
// One block per 16-message tile, 8 wave32s; all GEMMs via bf16 WMMA.
// All staging traffic (fmess tile, r1 tile, 96 gathered h rows) goes through
// the async global->LDS engine; one s_wait_asynccnt + barrier covers it.
__global__ __launch_bounds__(256)
void gru_step_kernel(const bf16_t* __restrict__ fmess_bf,
                     const int*    __restrict__ bgraph,
                     const bf16_t* __restrict__ r1_bf,
                     const bf16_t* __restrict__ Wzt, const float* __restrict__ bz,
                     const bf16_t* __restrict__ Urt, const float* __restrict__ br,
                     const bf16_t* __restrict__ Wht, const float* __restrict__ bh,
                     const bf16_t* __restrict__ h_in,
                     bf16_t*       __restrict__ h_out) {
    __shared__ bf16_t hnei[MAX_NB][16][HDIM];   // 24 KB
    __shared__ bf16_t fmt[16][FD_MESS];         //  5 KB
    __shared__ float  sumh_f[16][HDIM];         //  8 KB
    __shared__ bf16_t sumh_b[16][HDIM];         //  4 KB
    __shared__ bf16_t sumg_b[16][HDIM];         //  4 KB
    __shared__ bf16_t r1t[16][HDIM];            //  4 KB

    const int tid  = threadIdx.x;
    const int m0   = blockIdx.x * 16;
    const int lane = tid & 31;
    const int wave = tid >> 5;
    const int sub  = lane >> 4;     // which of 2 rows this wave-issue covers
    const int li   = lane & 15;     // 16B chunk within a 256B h row

    // Phase 1: async-stage fmess tile, r1 tile, and 96 gathered h rows.
    // fmess tile: 16x160 bf16 = 320 chunks of 16B (20 chunks/row, no split)
    for (int c = tid; c < 16 * FD_MESS / 8; c += 256)
        cp16_g2lds(fmess_bf + (size_t)m0 * FD_MESS + c * 8, &fmt[0][0] + c * 8);
    // r1 tile: 16x128 bf16 = 256 chunks of 16B
    cp16_g2lds(r1_bf + (size_t)m0 * HDIM + tid * 8, &r1t[0][0] + tid * 8);
    // neighbor gather: 96 (m,j) rows, 2 rows per wave-issue, 6 iterations
#pragma unroll
    for (int it = 0; it < 6; ++it) {
        int p  = it * 16 + wave * 2 + sub;
        int mi = p / MAX_NB, j = p % MAX_NB;
        int idx = bgraph[(size_t)(m0 + mi) * MAX_NB + j];
        cp16_g2lds(h_in + (size_t)idx * HDIM + li * 8, &hnei[j][mi][li * 8]);
    }
    async_fence();
    __syncthreads();

    // Phase 2: sum over neighbors
    for (int e = tid; e < 16 * HDIM; e += 256) {
        int row = e >> 7, c = e & 127;
        float s = 0.f;
#pragma unroll
        for (int j = 0; j < MAX_NB; ++j) s += (float)hnei[j][row][c];
        sumh_f[row][c] = s;
        sumh_b[row][c] = (bf16_t)s;
    }
    __syncthreads();

    // Phase 3: per-wave 16-col slice
    const int col = wave * 16 + (lane & 15);
    const int rl  = lane & 15, kh = lane >> 4;

    // z = sigmoid([fmess, sum_h] @ W_z + b_z)   (K = 160 + 128)
    v8f zacc = {0.f, 0.f, 0.f, 0.f, 0.f, 0.f, 0.f, 0.f};
#pragma unroll
    for (int kb = 0; kb < FD_MESS; kb += 32) {
        v16bf a = load_a_frag(&fmt[rl][kb], kh);
        zacc = wmma_bf16(a, load_b_frag(Wzt, FD_MESS + HDIM, col, kb, kh), zacc);
    }
#pragma unroll
    for (int kb = 0; kb < HDIM; kb += 32) {
        v16bf a = load_a_frag(&sumh_b[rl][kb], kh);
        zacc = wmma_bf16(a, load_b_frag(Wzt, FD_MESS + HDIM, col, FD_MESS + kb, kh), zacc);
    }
    const float bzc = bz[col];
    v8f zg;
#pragma unroll
    for (int i = 0; i < 8; ++i) zg[i] = sigmoid_f(zacc[i] + bzc);

    // r_j = sigmoid(r1 + h_nei_j @ U_r + b_r); sum_gated = sum_j r_j * h_nei_j
    const float brc = br[col];
    v8f sg = {0.f, 0.f, 0.f, 0.f, 0.f, 0.f, 0.f, 0.f};
#pragma unroll
    for (int j = 0; j < MAX_NB; ++j) {
        v8f racc = {0.f, 0.f, 0.f, 0.f, 0.f, 0.f, 0.f, 0.f};
#pragma unroll
        for (int kb = 0; kb < HDIM; kb += 32) {
            v16bf a = load_a_frag(&hnei[j][rl][kb], kh);
            racc = wmma_bf16(a, load_b_frag(Urt, HDIM, col, kb, kh), racc);
        }
#pragma unroll
        for (int i = 0; i < 8; ++i) {
            int row = i + 8 * kh;
            float rv = sigmoid_f(racc[i] + (float)r1t[row][col] + brc);
            sg[i] += rv * (float)hnei[j][row][col];
        }
    }
#pragma unroll
    for (int i = 0; i < 8; ++i) sumg_b[i + 8 * kh][col] = (bf16_t)sg[i];
    __syncthreads();

    // pre_h = tanh([fmess, sum_gated] @ W_h + b_h)
    v8f hacc = {0.f, 0.f, 0.f, 0.f, 0.f, 0.f, 0.f, 0.f};
#pragma unroll
    for (int kb = 0; kb < FD_MESS; kb += 32) {
        v16bf a = load_a_frag(&fmt[rl][kb], kh);
        hacc = wmma_bf16(a, load_b_frag(Wht, FD_MESS + HDIM, col, kb, kh), hacc);
    }
#pragma unroll
    for (int kb = 0; kb < HDIM; kb += 32) {
        v16bf a = load_a_frag(&sumg_b[rl][kb], kh);
        hacc = wmma_bf16(a, load_b_frag(Wht, FD_MESS + HDIM, col, FD_MESS + kb, kh), hacc);
    }
    const float bhc = bh[col];
#pragma unroll
    for (int i = 0; i < 8; ++i) {
        int row  = i + 8 * kh;
        int grow = m0 + row;
        float ph = tanhf(hacc[i] + bhc);
        float hn = (1.0f - zg[i]) * sumh_f[row][col] + zg[i] * ph;
        if (grow == 0) hn = 0.0f;  // message 0 is padding
        h_out[(size_t)grow * HDIM + col] = (bf16_t)hn;
    }
}

// --------------------------- node readout ----------------------------------
__global__ __launch_bounds__(256)
void node_out_kernel(const float* __restrict__ fnode,
                     const int*   __restrict__ agraph,
                     const bf16_t* __restrict__ h_fin,
                     const bf16_t* __restrict__ Wot, const float* __restrict__ bo,
                     float* __restrict__ out) {
    __shared__ bf16_t fnb[16][FD_NODE];         // 4 KB
    __shared__ bf16_t hnei[MAX_NB][16][HDIM];   // 24 KB
    __shared__ bf16_t nmsg[16][HDIM];           // 4 KB

    const int tid  = threadIdx.x;
    const int n0   = blockIdx.x * 16;
    const int lane = tid & 31;
    const int wave = tid >> 5;
    const int sub  = lane >> 4, li = lane & 15;

    // gather final h rows through the async engine
#pragma unroll
    for (int it = 0; it < 6; ++it) {
        int p  = it * 16 + wave * 2 + sub;
        int mi = p / MAX_NB, j = p % MAX_NB;
        int idx = agraph[(size_t)(n0 + mi) * MAX_NB + j];
        cp16_g2lds(h_fin + (size_t)idx * HDIM + li * 8, &hnei[j][mi][li * 8]);
    }
    // fnode tile: f32 -> bf16 (needs conversion, regular loads)
    for (int e = tid; e < 16 * FD_NODE; e += 256) {
        int r = e >> 7, c = e & 127;
        fnb[r][c] = (bf16_t)fnode[(size_t)(n0 + r) * FD_NODE + c];
    }
    async_fence();
    __syncthreads();

    for (int e = tid; e < 16 * HDIM; e += 256) {
        int row = e >> 7, c = e & 127;
        float s = 0.f;
#pragma unroll
        for (int j = 0; j < MAX_NB; ++j) s += (float)hnei[j][row][c];
        nmsg[row][c] = (bf16_t)s;
    }
    __syncthreads();

    const int col = wave * 16 + (lane & 15);
    const int rl  = lane & 15, kh = lane >> 4;

    v8f acc = {0.f, 0.f, 0.f, 0.f, 0.f, 0.f, 0.f, 0.f};
#pragma unroll
    for (int kb = 0; kb < FD_NODE; kb += 32) {
        v16bf a = load_a_frag(&fnb[rl][kb], kh);
        acc = wmma_bf16(a, load_b_frag(Wot, FD_NODE + HDIM, col, kb, kh), acc);
    }
#pragma unroll
    for (int kb = 0; kb < HDIM; kb += 32) {
        v16bf a = load_a_frag(&nmsg[rl][kb], kh);
        acc = wmma_bf16(a, load_b_frag(Wot, FD_NODE + HDIM, col, FD_NODE + kb, kh), acc);
    }
    const float bc = bo[col];
#pragma unroll
    for (int i = 0; i < 8; ++i) {
        int row = i + 8 * kh;
        int g   = n0 + row;
        float v = acc[i] + bc;
        v = v > 0.f ? v : 0.f;                  // ReLU
        if (g == 0) v = 0.f;                    // node 0 masked
        out[(size_t)g * HDIM + col] = v;
    }
}

// --------------------------- launch ----------------------------------------
extern "C" void kernel_launch(void* const* d_in, const int* in_sizes, int n_in,
                              void* d_out, int out_size, void* d_ws, size_t ws_size,
                              hipStream_t stream) {
    const float* fnode  = (const float*)d_in[0];
    const float* fmess  = (const float*)d_in[1];
    const int*   agraph = (const int*)d_in[2];
    const int*   bgraph = (const int*)d_in[3];
    const float* Wz = (const float*)d_in[4];
    const float* bz = (const float*)d_in[5];
    const float* Wr = (const float*)d_in[6];
    const float* Ur = (const float*)d_in[7];
    const float* br = (const float*)d_in[8];
    const float* Wh = (const float*)d_in[9];
    const float* bh = (const float*)d_in[10];
    const float* Wo = (const float*)d_in[11];
    const float* bo = (const float*)d_in[12];
    float* out = (float*)d_out;

    // workspace carve-up (~163.5 MB total)
    char* ws = (char*)d_ws;
    size_t off = 0;
    auto carve = [&](size_t bytes) -> void* {
        void* p = ws + off;
        off = (off + bytes + 255) & ~(size_t)255;
        return p;
    };
    bf16_t* fmess_bf = (bf16_t*)carve((size_t)N_MESS * FD_MESS * 2);  // 48.0 MB
    bf16_t* r1_bf    = (bf16_t*)carve((size_t)N_MESS * HDIM * 2);    // 38.4 MB
    bf16_t* h_a      = (bf16_t*)carve((size_t)N_MESS * HDIM * 2);    // 38.4 MB
    bf16_t* h_b      = (bf16_t*)carve((size_t)N_MESS * HDIM * 2);    // 38.4 MB
    bf16_t* Wzt = (bf16_t*)carve((size_t)(FD_MESS + HDIM) * 128 * 2);
    bf16_t* Wrt = (bf16_t*)carve((size_t)FD_MESS * 128 * 2);
    bf16_t* Urt = (bf16_t*)carve((size_t)HDIM * 128 * 2);
    bf16_t* Wht = (bf16_t*)carve((size_t)(FD_MESS + HDIM) * 128 * 2);
    bf16_t* Wot = (bf16_t*)carve((size_t)(FD_NODE + HDIM) * 128 * 2);

    // prep: transpose weights to bf16 [N][K]
    auto tgrid = [](int K) { return dim3((K * 128 + 255) / 256); };
    transpose_w_kernel<<<tgrid(FD_MESS + HDIM), 256, 0, stream>>>(Wz, Wzt, FD_MESS + HDIM);
    transpose_w_kernel<<<tgrid(FD_MESS),        256, 0, stream>>>(Wr, Wrt, FD_MESS);
    transpose_w_kernel<<<tgrid(HDIM),           256, 0, stream>>>(Ur, Urt, HDIM);
    transpose_w_kernel<<<tgrid(FD_MESS + HDIM), 256, 0, stream>>>(Wh, Wht, FD_MESS + HDIM);
    transpose_w_kernel<<<tgrid(FD_NODE + HDIM), 256, 0, stream>>>(Wo, Wot, FD_NODE + HDIM);

    // prep: fmess -> bf16, zero h_a
    {
        int n = N_MESS * FD_MESS;
        f32_to_bf16_kernel<<<(n + 255) / 256, 256, 0, stream>>>(fmess, fmess_bf, n);
        int m = N_MESS * HDIM;
        zero_bf16_kernel<<<(m + 255) / 256, 256, 0, stream>>>(h_a, m);
    }

    // loop-invariant r1 = fmess @ W_r
    r1_kernel<<<dim3(N_MESS / 16), 256, 0, stream>>>(fmess_bf, Wrt, r1_bf);

    // DEPTH GRU steps (ping-pong h buffers)
    bf16_t* hi = h_a;
    bf16_t* ho = h_b;
    for (int d = 0; d < DEPTH; ++d) {
        gru_step_kernel<<<dim3(N_MESS / 16), 256, 0, stream>>>(
            fmess_bf, bgraph, r1_bf, Wzt, bz, Urt, br, Wht, bh, hi, ho);
        bf16_t* t = hi; hi = ho; ho = t;
    }

    // node readout (final h is in `hi` after the last swap)
    node_out_kernel<<<dim3(N_NODES / 16), 256, 0, stream>>>(
        fnode, agraph, hi, Wot, bo, out);
}
